// CustomLayer_3710851744543
// MI455X (gfx1250) — matrix operand order repaired and checked by reading, not compiled, not run on hardware
//
#include <hip/hip_runtime.h>

// Problem constants (match reference).
#define KB_ROWS 8192           // B
#define KIN     2048           // IN_F
#define KOUT    2048           // OUT_F
#define KKT     (KIN / 32)     // 64 k-tiles of K=32

typedef __attribute__((ext_vector_type(16))) _Float16 v16h;
typedef __attribute__((ext_vector_type(8)))  float    v8f;
typedef __attribute__((ext_vector_type(2)))  _Float16 h2;

// Explicit global (addrspace 1) pointers so the backend emits global_load_* /
// global_store_* instead of flat_* (flat ties up DScnt and does per-lane
// aperture checks).
#define GAS __attribute__((address_space(1)))
template <typename T>
static __device__ __forceinline__ GAS T* gptr(T* p) {
  return (GAS T*)(unsigned long long)p;
}
template <typename T>
static __device__ __forceinline__ const GAS T* gptr(const T* p) {
  return (const GAS T*)(unsigned long long)p;
}

// WGP-scope prefetch (scope 0): pulls lines into ALL cache levels incl. WGP$.
// __builtin_prefetch only reaches SE/SYS scope, which per the ISA prefetch
// table does NOT populate the WGP cache -- useless when the stream is already
// GL2-resident. Two fixed immediates so BOTH unrolled bodies prefetch off the
// same (un-incremented) loop-carried pointer register: no temp v_add_nc_u64.
static __device__ __forceinline__ void prefetch_wgp_4k(const GAS _Float16* p) {
  asm volatile("global_prefetch_b8 %0, off offset:4096"
               :: "v"((unsigned long long)p));
}
static __device__ __forceinline__ void prefetch_wgp_5k(const GAS _Float16* p) {
  asm volatile("global_prefetch_b8 %0, off offset:5120"
               :: "v"((unsigned long long)p));
}

// ---------------------------------------------------------------------------
// Pack x [8192 x 2048, f32 row-major] into f16 WMMA A-fragments.
// A-fragment layout per CDNA5 ISA (16-bit A, 16x32 MxK, wave32):
//   lane l (l<16 -> grp0, l>=16 -> grp1), M = l%16
//   dword v (0..7): Klocal = (v>=4 ? 16:0) + (v%4)*2 + grp*8, holds K, K+1
// Buffer order: [mt][kt][lane][8 dwords]  (one 256-thread block per tile)
// ---------------------------------------------------------------------------
__global__ __launch_bounds__(256) void pack_x_frag(const float* __restrict__ x_,
                                                   h2* __restrict__ Af_) {
  const GAS float* x = gptr(x_);
  GAS h2* Af = gptr(Af_);
  const unsigned tid  = blockIdx.x * 256u + threadIdx.x;
  const unsigned t    = tid >> 8;
  const unsigned lane = (tid >> 3) & 31u;
  const unsigned v    = tid & 7u;
  const unsigned mt   = t / KKT;
  const unsigned kt   = t % KKT;
  const unsigned grp  = lane >> 4;
  const unsigned M    = mt * 16u + (lane & 15u);
  const unsigned Kl   = ((v >> 2) << 4) | ((v & 3u) << 1) | (grp << 3);
  const unsigned K    = kt * 32u + Kl;
  const size_t   base = (size_t)M * KIN + K;
  h2 o;
  o.x = (_Float16)x[base];
  o.y = (_Float16)x[base + 1u];
  Af[tid] = o;
}

// ---------------------------------------------------------------------------
// Fuse the Jeong IR-drop transform and pack into f16 WMMA B-fragments.
//   W_eff[i][j] = w[j][i] / (1 + (Rp/R_lrs) * seg(i,j) * w[j][i])
//   seg(i,j)    = (j+1) + (IN_F - i)
// B-fragment layout (16-bit B, 32x16 KxN, wave32):
//   lane l: N = l%16, grp = l/16; dword v (0..7): Klocal = (v>=4 ? 16:0) + (v%4)*2 + grp*8
// Buffer order: [nt][kt][lane][8 dwords]
// ---------------------------------------------------------------------------
__global__ __launch_bounds__(256) void pack_w_frag(const float* __restrict__ w_,
                                                   const float* __restrict__ Rp_p,
                                                   const float* __restrict__ Rlrs_p,
                                                   h2* __restrict__ Bf_) {
  const GAS float* w = gptr(w_);
  GAS h2* Bf = gptr(Bf_);
  const unsigned tid  = blockIdx.x * 256u + threadIdx.x;
  const unsigned t    = tid >> 8;
  const unsigned lane = (tid >> 3) & 31u;
  const unsigned v    = tid & 7u;
  const unsigned nt   = t / KKT;
  const unsigned kt   = t % KKT;
  const unsigned grp  = lane >> 4;
  const unsigned N    = nt * 16u + (lane & 15u);          // output col j
  const unsigned i0   = kt * 32u + ((v >> 2) << 4) + ((v & 3u) << 1) + (grp << 3); // input row i
  const float    c    = gptr(Rp_p)[0] / gptr(Rlrs_p)[0];

  const size_t base = (size_t)N * KIN + i0;
  const float w0 = w[base];
  const float w1 = w[base + 1u];
  const float seg0 = (float)(N + 1u) + (float)(KIN - i0);
  const float seg1 = seg0 - 1.0f;            // i increases by 1 -> seg drops by 1
  const float e0 = w0 / (1.0f + c * seg0 * w0);
  const float e1 = w1 / (1.0f + c * seg1 * w1);
  h2 o;
  o.x = (_Float16)e0;
  o.y = (_Float16)e1;
  Bf[tid] = o;
}

// ---------------------------------------------------------------------------
// WMMA GEMM: out[8192 x 2048] = A(8192x2048,f16 frags) * B(2048x2048,f16 frags) + bias
// Block = 256 thr = 8 waves; block tile 128x128; wave tile 64(M) x 32(N)
//   -> 4x2 grid of v_wmma_f32_16x16x32_f16 accumulators per wave.
// Manually unrolled x2 over k: both half-bodies load/prefetch off the same
// loop-carried pointers (offsets folded into the instructions), one +2 KB
// pointer bump per iteration. No LDS, no cross-lane shuffles; fragments are
// L2-resident and pushed into WGP$ by the WGP-scope prefetch stream.
// ---------------------------------------------------------------------------
__global__ __launch_bounds__(256) void gemm_wmma(const _Float16* __restrict__ Af_,
                                                 const _Float16* __restrict__ Bf_,
                                                 const float* __restrict__ bias_,
                                                 float* __restrict__ out_) {
  const GAS _Float16* Af = gptr(Af_);
  const GAS _Float16* Bf = gptr(Bf_);
  const GAS float* bias = gptr(bias_);
  GAS float* out = gptr(out_);

  const unsigned lane = threadIdx.x & 31u;
  const unsigned wave = threadIdx.x >> 5;
  const unsigned wm   = wave & 1u;          // 2 wave-rows
  const unsigned wn   = wave >> 1;          // 4 wave-cols
  const unsigned mt0  = blockIdx.y * 8u + wm * 4u;  // first of 4 M-tiles
  const unsigned nt0  = blockIdx.x * 8u + wn * 2u;  // first of 2 N-tiles

  v8f acc[4][2];
#pragma unroll
  for (int i = 0; i < 4; ++i)
#pragma unroll
    for (int j = 0; j < 2; ++j) acc[i][j] = (v8f)0.0f;

  // Fragment pointers: tile (mt,kt) lives at ((mt*KKT + kt)*32 + lane)*16 halves.
  const GAS _Float16* pa[4];
  const GAS _Float16* pb[2];
#pragma unroll
  for (int i = 0; i < 4; ++i)
    pa[i] = Af + ((size_t)(mt0 + i) * KKT * 32u + lane) * 16u;
#pragma unroll
  for (int j = 0; j < 2; ++j)
    pb[j] = Bf + ((size_t)(nt0 + j) * KKT * 32u + lane) * 16u;

#pragma unroll 1
  for (int kt = 0; kt < KKT; kt += 2) {
    v16h a0[4], a1[4];
    v16h b0[2], b1[2];
#pragma unroll
    for (int i = 0; i < 4; ++i) {
      a0[i] = *(const GAS v16h*)pa[i];              // kt   (offset 0)
      a1[i] = *(const GAS v16h*)(pa[i] + 512u);     // kt+1 (offset 1024B, folded)
      prefetch_wgp_4k(pa[i]);                       // kt+4 -> WGP$
      prefetch_wgp_5k(pa[i]);                       // kt+5 -> WGP$
    }
#pragma unroll
    for (int j = 0; j < 2; ++j) {
      b0[j] = *(const GAS v16h*)pb[j];
      b1[j] = *(const GAS v16h*)(pb[j] + 512u);
      prefetch_wgp_4k(pb[j]);
      prefetch_wgp_5k(pb[j]);
    }

#pragma unroll
    for (int i = 0; i < 4; ++i)
#pragma unroll
      for (int j = 0; j < 2; ++j)
        acc[i][j] = __builtin_amdgcn_wmma_f32_16x16x32_f16(
            false, a0[i], false, b0[j], (short)0, acc[i][j], false, false);
#pragma unroll
    for (int i = 0; i < 4; ++i)
#pragma unroll
      for (int j = 0; j < 2; ++j)
        acc[i][j] = __builtin_amdgcn_wmma_f32_16x16x32_f16(
            false, a1[i], false, b1[j], (short)0, acc[i][j], false, false);

#pragma unroll
    for (int i = 0; i < 4; ++i) pa[i] += 1024u;     // 2 k-tiles = 2048 bytes
#pragma unroll
    for (int j = 0; j < 2; ++j) pb[j] += 1024u;
  }

  // Epilogue per 32-bit C/D layout: VGPR r -> M = r (lanes 0-15) or r+8 (16-31),
  // N = lane % 16.
  const unsigned rsel = (lane >> 4) << 3;  // 0 or 8
  const unsigned ncol = lane & 15u;
#pragma unroll
  for (int j = 0; j < 2; ++j) {
    const unsigned Nc = (nt0 + j) * 16u + ncol;
    const float bv = bias[Nc];
#pragma unroll
    for (int i = 0; i < 4; ++i) {
      const unsigned Mb = (mt0 + i) * 16u + rsel;
#pragma unroll
      for (int r = 0; r < 8; ++r)
        out[(size_t)(Mb + r) * KOUT + Nc] = acc[i][j][r] + bv;
    }
  }
}

// ---------------------------------------------------------------------------
extern "C" void kernel_launch(void* const* d_in, const int* in_sizes, int n_in,
                              void* d_out, int out_size, void* d_ws, size_t ws_size,
                              hipStream_t stream) {
  const float* x    = (const float*)d_in[0];  // [8192, 2048]
  const float* w    = (const float*)d_in[1];  // [2048, 2048] (out, in)
  const float* bias = (const float*)d_in[2];  // [2048]
  const float* Rp   = (const float*)d_in[3];  // scalar
  const float* Rl   = (const float*)d_in[4];  // scalar
  float* out        = (float*)d_out;

  h2* Af = (h2*)d_ws;                                           // 33.55 MB
  h2* Bf = (h2*)((char*)d_ws + (size_t)KB_ROWS * KIN * 2u);     //  8.39 MB

  // One 256-thread block per 16x32 fragment tile.
  const unsigned nA = (KB_ROWS / 16) * KKT;   // 512 * 64 = 32768 blocks
  const unsigned nB = (KOUT    / 16) * KKT;   // 128 * 64 =  8192 blocks
  pack_x_frag<<<nA, 256, 0, stream>>>(x, Af);
  pack_w_frag<<<nB, 256, 0, stream>>>(w, Rp, Rl, Bf);

  dim3 grid(KOUT / 128, KB_ROWS / 128);       // (16, 64)
  gemm_wmma<<<grid, 256, 0, stream>>>((const _Float16*)Af, (const _Float16*)Bf,
                                      bias, out);
}
